// ContinuousThoughtMachine_48773648613662
// MI455X (gfx1250) — compile-verified, matrix operand order checked by reading, not run on hardware
//
#include <hip/hip_runtime.h>

typedef __attribute__((ext_vector_type(16))) __bf16 v16bf;
typedef __attribute__((ext_vector_type(8)))  __bf16 v8bf;
typedef __attribute__((ext_vector_type(8)))  float  v8f;

static constexpr int TB    = 256;
static constexpr int CB    = 64;      // batch
static constexpr int CS    = 196;     // kv tokens
static constexpr int CFEAT = 512;
static constexpr int CD    = 2048;    // d_model
static constexpr int CDIN  = 512;
static constexpr int CH    = 8;
static constexpr int CHD   = 64;
static constexpr int CM    = 25;
static constexpr int CHID  = 16;
static constexpr int CNS   = 512;
static constexpr int COUT  = 1000;
static constexpr int CT    = 50;
static constexpr int CBS   = CB * CS;   // 12544
static constexpr int CBD   = CB * CD;   // 131072

__device__ __forceinline__ unsigned short f2bf(float f) {
  __bf16 h = (__bf16)f;
  return __builtin_bit_cast(unsigned short, h);
}
__device__ __forceinline__ float sigmoidf_(float x) { return 1.f / (1.f + __expf(-x)); }

// ---------------------------------------------------------------------------
// Fragment loader: A row-major bf16, per-lane two 16B loads per K=32 step.
// Matches CDNA5 16-bit A layout: halfwords 0-7 = K kb..kb+7, 8-15 = K kb+16..kb+23
// with kb = (lane>>4)*8.
// ---------------------------------------------------------------------------
__device__ __forceinline__ v16bf load_frag(const __bf16* row, int k0, int kb) {
  v8bf a0 = *reinterpret_cast<const v8bf*>(row + k0 + kb);
  v8bf a1 = *reinterpret_cast<const v8bf*>(row + k0 + 16 + kb);
  return __builtin_shufflevector(a0, a1, 0,1,2,3,4,5,6,7,8,9,10,11,12,13,14,15);
}

// ---------------------------------------------------------------------------
// Generic bf16 WMMA GEMM: C(M,N) = A(M,K) * B(K,N) + bias(N)
// A: bf16 row-major (M x K); Bu: bf16 transposed (Npad x K, row-major).
// One wave -> 16x64 output tile via 4 accumulators of v_wmma_f32_16x16x32_bf16.
// ---------------------------------------------------------------------------
__global__ void __launch_bounds__(256)
k_gemm(const unsigned short* __restrict__ Au, const unsigned short* __restrict__ Bu,
       const float* __restrict__ bias, float* __restrict__ Cf,
       unsigned short* __restrict__ Cb, int Mdim, int Ndim, int Kdim, int ldc)
{
  const __bf16* A  = reinterpret_cast<const __bf16*>(Au);
  const __bf16* BT = reinterpret_cast<const __bf16*>(Bu);
  const int tilesN = (Ndim + 63) >> 6;
  const int tilesM = Mdim >> 4;
  const int gw = blockIdx.x * 8 + (threadIdx.x >> 5);
  if (gw >= tilesM * tilesN) return;           // wave-uniform: EXEC stays full
  const int mt = gw / tilesN, nt = gw % tilesN;
  const int lane = threadIdx.x & 31;
  const int lrow = lane & 15;
  const int kb   = (lane >> 4) << 3;

  const __bf16* Arow = A + (size_t)(mt * 16 + lrow) * Kdim;
  const __bf16* Br0  = BT + (size_t)(nt * 64 +  0 + lrow) * Kdim;
  const __bf16* Br1  = BT + (size_t)(nt * 64 + 16 + lrow) * Kdim;
  const __bf16* Br2  = BT + (size_t)(nt * 64 + 32 + lrow) * Kdim;
  const __bf16* Br3  = BT + (size_t)(nt * 64 + 48 + lrow) * Kdim;

  v8f acc0 = {}, acc1 = {}, acc2 = {}, acc3 = {};
  for (int k0 = 0; k0 < Kdim; k0 += 32) {
    v16bf af  = load_frag(Arow, k0, kb);
    v16bf bf0 = load_frag(Br0,  k0, kb);
    acc0 = __builtin_amdgcn_wmma_f32_16x16x32_bf16(false, af, false, bf0, (short)0, acc0, false, false);
    v16bf bf1 = load_frag(Br1,  k0, kb);
    acc1 = __builtin_amdgcn_wmma_f32_16x16x32_bf16(false, af, false, bf1, (short)0, acc1, false, false);
    v16bf bf2 = load_frag(Br2,  k0, kb);
    acc2 = __builtin_amdgcn_wmma_f32_16x16x32_bf16(false, af, false, bf2, (short)0, acc2, false, false);
    v16bf bf3 = load_frag(Br3,  k0, kb);
    acc3 = __builtin_amdgcn_wmma_f32_16x16x32_bf16(false, af, false, bf3, (short)0, acc3, false, false);
  }

  const int mbase = mt * 16 + ((lane >> 4) << 3);
  const int nbase = nt * 64 + (lane & 15);
#pragma unroll
  for (int v = 0; v < 8; ++v) {
    const int m = mbase + v;
    float vals[4] = {acc0[v], acc1[v], acc2[v], acc3[v]};
#pragma unroll
    for (int s = 0; s < 4; ++s) {
      const int n = nbase + s * 16;
      if (n < Ndim) {
        const float val = vals[s] + bias[n];
        const size_t ix = (size_t)m * ldc + n;
        if (Cf) Cf[ix] = val;
        if (Cb) Cb[ix] = f2bf(val);
      }
    }
  }
}

// ---------------------------------------------------------------------------
// Split-K bf16 WMMA GEMM (deterministic): partial[ks] = A[:, ks-chunk] * B.
// Requires M%16==0, N%64==0, (K/ksplit)%32==0. No bias, raw fp32 partials.
// Boosts wave count 8x for the M=64 synapse GEMM (occupancy on MI455X WGPs).
// ---------------------------------------------------------------------------
__global__ void __launch_bounds__(256)
k_gemm_sk(const unsigned short* __restrict__ Au, const unsigned short* __restrict__ Bu,
          float* __restrict__ Cpart, int Mdim, int Ndim, int Kdim, int ksplit)
{
  const __bf16* A  = reinterpret_cast<const __bf16*>(Au);
  const __bf16* BT = reinterpret_cast<const __bf16*>(Bu);
  const int tilesN = Ndim >> 6;
  const int tilesM = Mdim >> 4;
  const int gw = blockIdx.x * 8 + (threadIdx.x >> 5);
  if (gw >= tilesM * tilesN * ksplit) return;  // wave-uniform
  const int ks  = gw / (tilesM * tilesN);
  const int rem = gw - ks * tilesM * tilesN;
  const int mt = rem / tilesN, nt = rem % tilesN;
  const int kchunk = Kdim / ksplit;
  const int kbeg = ks * kchunk, kend = kbeg + kchunk;
  const int lane = threadIdx.x & 31;
  const int lrow = lane & 15;
  const int kb   = (lane >> 4) << 3;

  const __bf16* Arow = A + (size_t)(mt * 16 + lrow) * Kdim;
  const __bf16* Br0  = BT + (size_t)(nt * 64 +  0 + lrow) * Kdim;
  const __bf16* Br1  = BT + (size_t)(nt * 64 + 16 + lrow) * Kdim;
  const __bf16* Br2  = BT + (size_t)(nt * 64 + 32 + lrow) * Kdim;
  const __bf16* Br3  = BT + (size_t)(nt * 64 + 48 + lrow) * Kdim;

  v8f acc0 = {}, acc1 = {}, acc2 = {}, acc3 = {};
  for (int k0 = kbeg; k0 < kend; k0 += 32) {
    v16bf af  = load_frag(Arow, k0, kb);
    v16bf bf0 = load_frag(Br0,  k0, kb);
    acc0 = __builtin_amdgcn_wmma_f32_16x16x32_bf16(false, af, false, bf0, (short)0, acc0, false, false);
    v16bf bf1 = load_frag(Br1,  k0, kb);
    acc1 = __builtin_amdgcn_wmma_f32_16x16x32_bf16(false, af, false, bf1, (short)0, acc1, false, false);
    v16bf bf2 = load_frag(Br2,  k0, kb);
    acc2 = __builtin_amdgcn_wmma_f32_16x16x32_bf16(false, af, false, bf2, (short)0, acc2, false, false);
    v16bf bf3 = load_frag(Br3,  k0, kb);
    acc3 = __builtin_amdgcn_wmma_f32_16x16x32_bf16(false, af, false, bf3, (short)0, acc3, false, false);
  }

  float* P = Cpart + (size_t)ks * Mdim * Ndim;
  const int mbase = mt * 16 + ((lane >> 4) << 3);
  const int nbase = nt * 64 + (lane & 15);
#pragma unroll
  for (int v = 0; v < 8; ++v) {
    const size_t mrow = (size_t)(mbase + v) * Ndim + nbase;
    P[mrow +  0] = acc0[v];
    P[mrow + 16] = acc1[v];
    P[mrow + 32] = acc2[v];
    P[mrow + 48] = acc3[v];
  }
}

// Sum split-K partials + bias; store fp32 and/or bf16 with leading dim ldc.
__global__ void k_reduce_bias(const float* __restrict__ Cpart, const float* __restrict__ bias,
                              float* __restrict__ Cf, unsigned short* __restrict__ Cb,
                              int Mdim, int Ndim, int ldc, int nsplit)
{
  const int idx = blockIdx.x * TB + threadIdx.x;
  if (idx >= Mdim * Ndim) return;
  const int m = idx / Ndim, n = idx - m * Ndim;
  float s = bias[n];
  for (int k = 0; k < nsplit; ++k) s += Cpart[(size_t)k * Mdim * Ndim + idx];
  const size_t ix = (size_t)m * ldc + n;
  if (Cf) Cf[ix] = s;
  if (Cb) Cb[ix] = f2bf(s);
}

// ---------------------------------------------------------------------------
// Helpers: conversions / transposes / decay rates
// ---------------------------------------------------------------------------
__global__ void k_convert_bf(const float* __restrict__ in, unsigned short* __restrict__ out, int n) {
  int i = blockIdx.x * TB + threadIdx.x;
  if (i < n) out[i] = f2bf(in[i]);
}

// in: (K x N) fp32 row-major  ->  out: (Npad x K) bf16 row-major, zero pad
__global__ void k_transpose_bf(const float* __restrict__ in, unsigned short* __restrict__ out,
                               int K, int N, int Npad) {
  int idx = blockIdx.x * TB + threadIdx.x;
  if (idx >= Npad * K) return;
  int n = idx / K, k = idx - n * K;
  float v = (n < N) ? in[(size_t)k * N + n] : 0.f;
  out[idx] = f2bf(v);
}

__global__ void k_decay(const float* __restrict__ d, float* __restrict__ r, int n) {
  int i = blockIdx.x * TB + threadIdx.x;
  if (i < n) r[i] = __expf(-fminf(fmaxf(d[i], 0.f), 15.f));
}

// ---------------------------------------------------------------------------
// Row LayerNorm (fp32 in, bf16 out), one block per row
// ---------------------------------------------------------------------------
__global__ void __launch_bounds__(256)
k_ln_bf(const float* __restrict__ in, const float* __restrict__ g,
        const float* __restrict__ bt, unsigned short* __restrict__ out, int N)
{
  int row = blockIdx.x, tid = threadIdx.x;
  const float* x = in + (size_t)row * N;
  __shared__ float red[256];
  float s = 0.f;
  for (int i = tid; i < N; i += 256) s += x[i];
  red[tid] = s; __syncthreads();
  for (int o = 128; o > 0; o >>= 1) { if (tid < o) red[tid] += red[tid + o]; __syncthreads(); }
  float mean = red[0] / N; __syncthreads();
  float v = 0.f;
  for (int i = tid; i < N; i += 256) { float d = x[i] - mean; v += d * d; }
  red[tid] = v; __syncthreads();
  for (int o = 128; o > 0; o >>= 1) { if (tid < o) red[tid] += red[tid + o]; __syncthreads(); }
  float inv = rsqrtf(red[0] / N + 1e-5f);
  unsigned short* y = out + (size_t)row * N;
  for (int i = tid; i < N; i += 256) y[i] = f2bf((x[i] - mean) * inv * g[i] + bt[i]);
}

// ---------------------------------------------------------------------------
// Attention: one block per (b,h); q (fp32), kh/vh (fp32) -> o (bf16)
// ---------------------------------------------------------------------------
__global__ void __launch_bounds__(256)
k_attn(const float* __restrict__ qh, const float* __restrict__ kh,
       const float* __restrict__ vh, unsigned short* __restrict__ obf)
{
  const int bh = blockIdx.x;
  const int b = bh >> 3, h = bh & 7;
  const int tid = threadIdx.x;
  __shared__ float qs[CHD];
  __shared__ float sc[256];
  __shared__ float red[256];
  if (tid < CHD) qs[tid] = qh[(size_t)b * CDIN + h * CHD + tid];
  __syncthreads();
  float s = -1e30f;
  if (tid < CS) {
    const float* kp = kh + (size_t)(b * CS + tid) * CDIN + h * CHD;
    float d = 0.f;
#pragma unroll 8
    for (int i = 0; i < CHD; ++i) d += qs[i] * kp[i];
    s = d * 0.125f;                      // 1/sqrt(64)
  }
  red[tid] = s; __syncthreads();
  for (int o = 128; o > 0; o >>= 1) { if (tid < o) red[tid] = fmaxf(red[tid], red[tid + o]); __syncthreads(); }
  const float mx = red[0]; __syncthreads();
  const float e = (tid < CS) ? __expf(s - mx) : 0.f;
  sc[tid] = e; red[tid] = e; __syncthreads();
  for (int o = 128; o > 0; o >>= 1) { if (tid < o) red[tid] += red[tid + o]; __syncthreads(); }
  const float inv = 1.f / red[0];
  __syncthreads();
  if (tid < CHD) {
    float acc = 0.f;
    for (int si = 0; si < CS; ++si)
      acc += sc[si] * vh[(size_t)(b * CS + si) * CDIN + h * CHD + tid];
    obf[(size_t)b * CDIN + h * CHD + tid] = f2bf(acc * inv);
  }
}

// ---------------------------------------------------------------------------
// GLU + LayerNorm over D, write into trace ring slot. One block per batch row.
// ---------------------------------------------------------------------------
__global__ void __launch_bounds__(256)
k_glu_ln_ring(const float* __restrict__ syn, const float* __restrict__ g,
              const float* __restrict__ bt, float* __restrict__ ring, int slot)
{
  const int b = blockIdx.x, tid = threadIdx.x;
  const float* row = syn + (size_t)b * (2 * CD);
  __shared__ float gg[CD];
  __shared__ float red[256];
  for (int j = tid; j < CD; j += 256) gg[j] = row[j] * sigmoidf_(row[CD + j]);
  __syncthreads();
  float s = 0.f;
  for (int j = tid; j < CD; j += 256) s += gg[j];
  red[tid] = s; __syncthreads();
  for (int o = 128; o > 0; o >>= 1) { if (tid < o) red[tid] += red[tid + o]; __syncthreads(); }
  const float mean = red[0] * (1.f / CD); __syncthreads();
  float v = 0.f;
  for (int j = tid; j < CD; j += 256) { float dd = gg[j] - mean; v += dd * dd; }
  red[tid] = v; __syncthreads();
  for (int o = 128; o > 0; o >>= 1) { if (tid < o) red[tid] += red[tid + o]; __syncthreads(); }
  const float inv = rsqrtf(red[0] * (1.f / CD) + 1e-5f);
  float* dst = ring + (size_t)slot * CBD + (size_t)b * CD;
  for (int j = tid; j < CD; j += 256) dst[j] = (gg[j] - mean) * inv * g[j] + bt[j];
}

// ---------------------------------------------------------------------------
// Neuron-level model: per (b,d) GLU MLP over M=25 history. head = ring start.
// ---------------------------------------------------------------------------
__global__ void __launch_bounds__(256)
k_nlm(const float* __restrict__ ring, int head,
      const float* __restrict__ w1, const float* __restrict__ b1,
      const float* __restrict__ w2, const float* __restrict__ b2,
      float* __restrict__ act, unsigned short* __restrict__ prebf)
{
  const int idx = blockIdx.x * TB + threadIdx.x;
  if (idx >= CBD) return;
  const int b = idx >> 11, d = idx & (CD - 1);
  float tr[CM];
#pragma unroll
  for (int m = 0; m < CM; ++m) {
    int p = head + m; if (p >= CM) p -= CM;
    tr[m] = ring[(size_t)p * CBD + idx];
  }
  float p0 = b2[d * 2 + 0], p1 = b2[d * 2 + 1];
  for (int h = 0; h < CHID; ++h) {
    float u = b1[d * 2 * CHID + h];
    float gte = b1[d * 2 * CHID + CHID + h];
#pragma unroll
    for (int m = 0; m < CM; ++m) {
      const float t = tr[m];
      u   += t * w1[((size_t)m * 2 * CHID + h) * CD + d];
      gte += t * w1[((size_t)m * 2 * CHID + CHID + h) * CD + d];
    }
    const float h1 = u * sigmoidf_(gte);
    p0 += h1 * w2[(size_t)(h * 2 + 0) * CD + d];
    p1 += h1 * w2[(size_t)(h * 2 + 1) * CD + d];
  }
  const float a = p0 * sigmoidf_(p1);
  act[idx] = a;
  prebf[(size_t)b * (2 * CD) + CD + d] = f2bf(a);
}

// ---------------------------------------------------------------------------
// Synchronisation update: a = r*a + act[il]*act[ir]; b = r*b + 1; out bf16
// ---------------------------------------------------------------------------
__global__ void k_sync(const float* __restrict__ act, const int* __restrict__ il,
                       const int* __restrict__ ir, const float* __restrict__ r,
                       float* __restrict__ aS, float* __restrict__ bS,
                       unsigned short* __restrict__ syncb)
{
  const int idx = blockIdx.x * TB + threadIdx.x;
  if (idx >= CB * CNS) return;
  const int b = idx >> 9, j = idx & (CNS - 1);
  const float pa = act[(size_t)b * CD + il[j]] * act[(size_t)b * CD + ir[j]];
  const float rr = r[j];
  const float a  = rr * aS[idx] + pa;
  const float bb = rr * bS[idx] + 1.f;
  aS[idx] = a; bS[idx] = bb;
  syncb[idx] = f2bf(a * rsqrtf(bb));
}

// ---------------------------------------------------------------------------
// State initialization
// ---------------------------------------------------------------------------
__global__ void k_init_state(const float* __restrict__ start_act,
                             const float* __restrict__ start_trace,
                             float* __restrict__ act, unsigned short* __restrict__ prebf,
                             float* __restrict__ ring)
{
  const int idx = blockIdx.x * TB + threadIdx.x;
  if (idx >= CBD) return;
  const int b = idx >> 11, d = idx & (CD - 1);
  const float a = start_act[d];
  act[idx] = a;
  prebf[(size_t)b * (2 * CD) + CD + d] = f2bf(a);
#pragma unroll
  for (int m = 0; m < CM; ++m)
    ring[(size_t)m * CBD + idx] = start_trace[d * CM + m];
}

__global__ void k_init_sync(const float* __restrict__ start_act,
                            const int* __restrict__ ilo, const int* __restrict__ iro,
                            float* __restrict__ aA, float* __restrict__ bA,
                            float* __restrict__ aO, float* __restrict__ bO)
{
  const int idx = blockIdx.x * TB + threadIdx.x;
  if (idx >= CB * CNS) return;
  const int j = idx & (CNS - 1);
  aA[idx] = 0.f; bA[idx] = 0.f;
  aO[idx] = start_act[ilo[j]] * start_act[iro[j]];
  bO[idx] = 1.f;
}

// ---------------------------------------------------------------------------
// Output: write predictions (strided) + normalized-entropy certainties
// ---------------------------------------------------------------------------
__global__ void __launch_bounds__(256)
k_out(const float* __restrict__ pred, float* __restrict__ out, int t)
{
  const int b = blockIdx.x, tid = threadIdx.x;
  const float* p = pred + (size_t)b * COUT;
  __shared__ float red[256];
  float mx = -1e30f;
  for (int i = tid; i < COUT; i += 256) mx = fmaxf(mx, p[i]);
  red[tid] = mx; __syncthreads();
  for (int o = 128; o > 0; o >>= 1) { if (tid < o) red[tid] = fmaxf(red[tid], red[tid + o]); __syncthreads(); }
  mx = red[0]; __syncthreads();
  float se = 0.f;
  for (int i = tid; i < COUT; i += 256) se += __expf(p[i] - mx);
  red[tid] = se; __syncthreads();
  for (int o = 128; o > 0; o >>= 1) { if (tid < o) red[tid] += red[tid + o]; __syncthreads(); }
  const float lse = mx + __logf(red[0]); __syncthreads();
  float ent = 0.f;
  for (int i = tid; i < COUT; i += 256) { float lp = p[i] - lse; ent += __expf(lp) * lp; }
  red[tid] = ent; __syncthreads();
  for (int o = 128; o > 0; o >>= 1) { if (tid < o) red[tid] += red[tid + o]; __syncthreads(); }
  if (tid == 0) {
    const float ne = -red[0] * (1.f / 6.907755278982137f);   // / log(1000)
    const size_t cbase = (size_t)CB * COUT * CT + (size_t)b * 2 * CT;
    out[cbase + t] = ne;
    out[cbase + CT + t] = 1.f - ne;
  }
  const size_t pbase = (size_t)b * COUT * CT;
  for (int i = tid; i < COUT; i += 256) out[pbase + (size_t)i * CT + t] = p[i];
}

// ---------------------------------------------------------------------------
// Host driver
// ---------------------------------------------------------------------------
static inline int gdiv(long long a, int b) { return (int)((a + b - 1) / b); }

extern "C" void kernel_launch(void* const* d_in, const int* in_sizes, int n_in,
                              void* d_out, int out_size, void* d_ws, size_t ws_size,
                              hipStream_t stream)
{
  (void)in_sizes; (void)n_in; (void)out_size; (void)ws_size;
  const float* x         = (const float*)d_in[0];
  const float* kv_w      = (const float*)d_in[1];
  const float* kv_b      = (const float*)d_in[2];
  const float* kv_ln_g   = (const float*)d_in[3];
  const float* kv_ln_b   = (const float*)d_in[4];
  const float* q_w       = (const float*)d_in[5];
  const float* q_b       = (const float*)d_in[6];
  const float* attn_wq   = (const float*)d_in[7];
  const float* attn_bq   = (const float*)d_in[8];
  const float* attn_wk   = (const float*)d_in[9];
  const float* attn_bk   = (const float*)d_in[10];
  const float* attn_wv   = (const float*)d_in[11];
  const float* attn_bv   = (const float*)d_in[12];
  const float* attn_wo   = (const float*)d_in[13];
  const float* attn_bo   = (const float*)d_in[14];
  const float* ro_w      = (const float*)d_in[15];
  const float* ro_b      = (const float*)d_in[16];
  const float* syn_w     = (const float*)d_in[17];
  const float* syn_b     = (const float*)d_in[18];
  const float* syn_ln_g  = (const float*)d_in[19];
  const float* syn_ln_b  = (const float*)d_in[20];
  const float* nlm_w1    = (const float*)d_in[21];
  const float* nlm_b1    = (const float*)d_in[22];
  const float* nlm_w2    = (const float*)d_in[23];
  const float* nlm_b2    = (const float*)d_in[24];
  const float* start_trace = (const float*)d_in[25];
  const float* start_act   = (const float*)d_in[26];
  const float* decay_action= (const float*)d_in[27];
  const float* decay_out   = (const float*)d_in[28];
  const float* out_w     = (const float*)d_in[29];
  const float* out_b     = (const float*)d_in[30];
  const int* idx_la = (const int*)d_in[31];
  const int* idx_ra = (const int*)d_in[32];
  const int* idx_lo = (const int*)d_in[33];
  const int* idx_ro = (const int*)d_in[34];
  float* outp = (float*)d_out;

  char* ws = (char*)d_ws;
  size_t off = 0;
  auto alloc = [&](size_t bytes) -> void* {
    void* p = ws + off;
    off = (off + bytes + 255) & ~(size_t)255;
    return p;
  };

  unsigned short* xbf  = (unsigned short*)alloc((size_t)CBS * CFEAT * 2);
  unsigned short* wKV  = (unsigned short*)alloc((size_t)512 * 512 * 2);
  unsigned short* wQ   = (unsigned short*)alloc((size_t)512 * 512 * 2);
  unsigned short* wWQ  = (unsigned short*)alloc((size_t)512 * 512 * 2);
  unsigned short* wWK  = (unsigned short*)alloc((size_t)512 * 512 * 2);
  unsigned short* wWV  = (unsigned short*)alloc((size_t)512 * 512 * 2);
  unsigned short* wWO  = (unsigned short*)alloc((size_t)512 * 512 * 2);
  unsigned short* wRO  = (unsigned short*)alloc((size_t)2048 * 512 * 2);
  unsigned short* wSYN = (unsigned short*)alloc((size_t)4096 * 4096 * 2);
  unsigned short* wOUT = (unsigned short*)alloc((size_t)1024 * 512 * 2);   // padded 1000->1024
  float*          kvpre= (float*)alloc((size_t)CBS * 512 * 4);
  unsigned short* kvbf = (unsigned short*)alloc((size_t)CBS * 512 * 2);
  float*          khf  = (float*)alloc((size_t)CBS * 512 * 4);
  float*          vhf  = (float*)alloc((size_t)CBS * 512 * 4);
  float*          rA   = (float*)alloc((size_t)512 * 4);
  float*          rO   = (float*)alloc((size_t)512 * 4);
  float*          actf = (float*)alloc((size_t)CBD * 4);
  float*          ring = (float*)alloc((size_t)CM * CBD * 4);
  float*          aA   = (float*)alloc((size_t)CB * CNS * 4);
  float*          bA   = (float*)alloc((size_t)CB * CNS * 4);
  float*          aO   = (float*)alloc((size_t)CB * CNS * 4);
  float*          bO   = (float*)alloc((size_t)CB * CNS * 4);
  unsigned short* sAb  = (unsigned short*)alloc((size_t)CB * CNS * 2);
  unsigned short* sOb  = (unsigned short*)alloc((size_t)CB * CNS * 2);
  unsigned short* qbf  = (unsigned short*)alloc((size_t)CB * 512 * 2);
  float*          qhf  = (float*)alloc((size_t)CB * 512 * 4);
  unsigned short* obf  = (unsigned short*)alloc((size_t)CB * 512 * 2);
  unsigned short* oobf = (unsigned short*)alloc((size_t)CB * 512 * 2);
  unsigned short* prebf= (unsigned short*)alloc((size_t)CB * 4096 * 2);
  float*          synf = (float*)alloc((size_t)CB * 4096 * 4);
  float*          predf= (float*)alloc((size_t)CB * COUT * 4);
  float*          part = (float*)alloc((size_t)8 * CB * 4096 * 4);       // split-K partials

  auto gemm = [&](const unsigned short* A, const unsigned short* BT, const float* bias,
                  float* Cf, unsigned short* Cb, int M, int N, int K, int ldc) {
    const int tiles = (M / 16) * ((N + 63) / 64);
    k_gemm<<<dim3(gdiv(tiles, 8)), dim3(256), 0, stream>>>(A, BT, bias, Cf, Cb, M, N, K, ldc);
  };
  auto gemm_sk = [&](const unsigned short* A, const unsigned short* BT,
                     int M, int N, int K, int ksplit) {
    const int tiles = (M / 16) * (N / 64) * ksplit;
    k_gemm_sk<<<dim3(gdiv(tiles, 8)), dim3(256), 0, stream>>>(A, BT, part, M, N, K, ksplit);
  };

  // ---- one-time precompute (per launch, deterministic) ----
  k_convert_bf<<<dim3(gdiv((long long)CBS * CFEAT, TB)), dim3(TB), 0, stream>>>(x, xbf, CBS * CFEAT);
  k_transpose_bf<<<dim3(gdiv(512 * 512, TB)), dim3(TB), 0, stream>>>(kv_w,    wKV, 512, 512, 512);
  k_transpose_bf<<<dim3(gdiv(512 * 512, TB)), dim3(TB), 0, stream>>>(q_w,     wQ,  512, 512, 512);
  k_transpose_bf<<<dim3(gdiv(512 * 512, TB)), dim3(TB), 0, stream>>>(attn_wq, wWQ, 512, 512, 512);
  k_transpose_bf<<<dim3(gdiv(512 * 512, TB)), dim3(TB), 0, stream>>>(attn_wk, wWK, 512, 512, 512);
  k_transpose_bf<<<dim3(gdiv(512 * 512, TB)), dim3(TB), 0, stream>>>(attn_wv, wWV, 512, 512, 512);
  k_transpose_bf<<<dim3(gdiv(512 * 512, TB)), dim3(TB), 0, stream>>>(attn_wo, wWO, 512, 512, 512);
  k_transpose_bf<<<dim3(gdiv(2048 * 512, TB)), dim3(TB), 0, stream>>>(ro_w,  wRO, 512, 2048, 2048);
  k_transpose_bf<<<dim3(gdiv((long long)4096 * 4096, TB)), dim3(TB), 0, stream>>>(syn_w, wSYN, 4096, 4096, 4096);
  k_transpose_bf<<<dim3(gdiv(1024 * 512, TB)), dim3(TB), 0, stream>>>(out_w, wOUT, 512, 1000, 1024);
  k_decay<<<dim3(gdiv(512, TB)), dim3(TB), 0, stream>>>(decay_action, rA, 512);
  k_decay<<<dim3(gdiv(512, TB)), dim3(TB), 0, stream>>>(decay_out,    rO, 512);

  gemm(xbf, wKV, kv_b, kvpre, nullptr, CBS, 512, 512, 512);            // x @ kv_w + b
  k_ln_bf<<<dim3(CBS), dim3(256), 0, stream>>>(kvpre, kv_ln_g, kv_ln_b, kvbf, 512);
  gemm(kvbf, wWK, attn_bk, khf, nullptr, CBS, 512, 512, 512);          // kh
  gemm(kvbf, wWV, attn_bv, vhf, nullptr, CBS, 512, 512, 512);          // vh

  k_init_state<<<dim3(gdiv(CBD, TB)), dim3(TB), 0, stream>>>(start_act, start_trace, actf, prebf, ring);
  k_init_sync<<<dim3(gdiv(CB * CNS, TB)), dim3(TB), 0, stream>>>(start_act, idx_lo, idx_ro, aA, bA, aO, bO);

  // ---- recurrent loop: 50 internal ticks ----
  for (int t = 0; t < CT; ++t) {
    k_sync<<<dim3(gdiv(CB * CNS, TB)), dim3(TB), 0, stream>>>(actf, idx_la, idx_ra, rA, aA, bA, sAb);
    gemm(sAb,  wQ,  q_b,     nullptr, qbf,  CB, 512, 512, 512);        // q = sync_a @ q_w
    gemm(qbf,  wWQ, attn_bq, qhf,     nullptr, CB, 512, 512, 512);     // qh
    k_attn<<<dim3(CB * CH), dim3(256), 0, stream>>>(qhf, khf, vhf, obf);
    gemm(obf,  wWO, attn_bo, nullptr, oobf, CB, 512, 512, 512);        // o @ wo

    // readout: split-K x4 (128 -> 512 waves), reduce into pre[:, :2048] (bf16)
    gemm_sk(oobf, wRO, CB, 2048, 512, 4);
    k_reduce_bias<<<dim3(gdiv(CB * 2048, TB)), dim3(TB), 0, stream>>>(
        part, ro_b, nullptr, prebf, CB, 2048, 4096, 4);

    // synapse: split-K x8 (256 -> 2048 waves), reduce to fp32
    gemm_sk(prebf, wSYN, CB, 4096, 4096, 8);
    k_reduce_bias<<<dim3(gdiv(CB * 4096, TB)), dim3(TB), 0, stream>>>(
        part, syn_b, synf, nullptr, CB, 4096, 4096, 8);

    k_glu_ln_ring<<<dim3(CB), dim3(256), 0, stream>>>(synf, syn_ln_g, syn_ln_b, ring, t % CM);
    k_nlm<<<dim3(gdiv(CBD, TB)), dim3(TB), 0, stream>>>(ring, (t + 1) % CM,
                                                        nlm_w1, nlm_b1, nlm_w2, nlm_b2, actf, prebf);
    k_sync<<<dim3(gdiv(CB * CNS, TB)), dim3(TB), 0, stream>>>(actf, idx_lo, idx_ro, rO, aO, bO, sOb);
    gemm(sOb, wOUT, out_b, predf, nullptr, CB, 1000, 512, 1000);       // prediction
    k_out<<<dim3(CB), dim3(256), 0, stream>>>(predf, outp, t);
  }
}
